// DiT_29025388986539
// MI455X (gfx1250) — matrix operand order repaired and checked by reading, not compiled, not run on hardware
//
#include <hip/hip_runtime.h>
#include <hip/hip_bf16.h>
#include <cstdint>
#include <cstddef>

// ---------------------------------------------------------------------------
// DiT forward for MI455X (gfx1250, wave32, WMMA bf16 16x16x32, f32 accum)
// ---------------------------------------------------------------------------

typedef __attribute__((ext_vector_type(16))) __bf16 v16bf;
typedef __attribute__((ext_vector_type(8)))  float  v8f;

#define DEV static __device__ __forceinline__

// Problem constants
static constexpr int BATCH = 16;
static constexpr int HID   = 768;
static constexpr int HEADS = 12;
static constexpr int HD    = 64;
static constexpr int DEPTH = 12;
static constexpr int MLPD  = 3072;
static constexpr int NTOK  = 1024;   // 32x32 patches
static constexpr int MTOT  = BATCH * NTOK; // 16384
static constexpr int PATCHK = 192;   // 3*8*8
static constexpr int CCN   = 6 * HID; // 4608

// Native bf16 convert (fptrunc, RNE)
DEV unsigned short f2bf(float f) {
  union { __bf16 h; unsigned short u; } cv;
  cv.h = (__bf16)f;
  return cv.u;
}

union FragU { v16bf v; uint4 q[2]; unsigned short s[16]; };

DEV v8f v8f_zero() { v8f z = {0.f,0.f,0.f,0.f,0.f,0.f,0.f,0.f}; return z; }

// max-reduce across a 16-lane row via DPP16 butterfly (no LDS traffic)
DEV float wave16_max(float x) {
  union { float f; int i; } a, b;
  a.f = x;
  b.i = __builtin_amdgcn_update_dpp(0, a.i, 0xB1, 0xF, 0xF, true);  // quad_perm [1,0,3,2]
  a.f = fmaxf(a.f, b.f);
  b.i = __builtin_amdgcn_update_dpp(0, a.i, 0x4E, 0xF, 0xF, true);  // quad_perm [2,3,0,1]
  a.f = fmaxf(a.f, b.f);
  b.i = __builtin_amdgcn_update_dpp(0, a.i, 0x141, 0xF, 0xF, true); // row_half_mirror
  a.f = fmaxf(a.f, b.f);
  b.i = __builtin_amdgcn_update_dpp(0, a.i, 0x140, 0xF, 0xF, true); // row_mirror
  a.f = fmaxf(a.f, b.f);
  return a.f;
}

// A operand (16x32, MxK): lane<16 -> row=lane, K {k0..k0+15} contiguous
//                         lane>=16 -> same row, K {k0+16..k0+31}
DEV v16bf fragA_bf16(const unsigned short* base, int ld, int row0, int k0, int lane) {
  int r  = row0 + (lane & 15);
  int ks = k0 + ((lane >> 4) << 4);
  const unsigned short* p = base + (size_t)r * ld + ks;
  FragU f;
  f.q[0] = *(const uint4*)(p);
  f.q[1] = *(const uint4*)(p + 8);
  return f.v;
}

// B operand (32x16, KxN): lane<16 -> col=lane, K {k0..k0+15} contiguous
//                         lane>=16 -> same col, K {k0+16..k0+31}
DEV v16bf fragB_bf16(const unsigned short* base, int ld, int col0, int k0, int lane) {
  int n  = col0 + (lane & 15);
  int ks = k0 + ((lane >> 4) << 4);
  const unsigned short* p = base + (size_t)n * ld + ks;
  FragU f;
  f.q[0] = *(const uint4*)(p);
  f.q[1] = *(const uint4*)(p + 8);
  return f.v;
}

// B operand built on the fly from f32 weights (row n of W is K-contiguous)
DEV v16bf fragB_f32(const float* base, int ld, int col0, int k0, int lane) {
  int n  = col0 + (lane & 15);
  int ks = k0 + ((lane >> 4) << 4);
  const float* p = base + (size_t)n * ld + ks;
  float4 x0 = *(const float4*)(p);
  float4 x1 = *(const float4*)(p + 4);
  float4 x2 = *(const float4*)(p + 8);
  float4 x3 = *(const float4*)(p + 12);
  FragU f;
  f.v[0]  = (__bf16)x0.x; f.v[1]  = (__bf16)x0.y; f.v[2]  = (__bf16)x0.z; f.v[3]  = (__bf16)x0.w;
  f.v[4]  = (__bf16)x1.x; f.v[5]  = (__bf16)x1.y; f.v[6]  = (__bf16)x1.z; f.v[7]  = (__bf16)x1.w;
  f.v[8]  = (__bf16)x2.x; f.v[9]  = (__bf16)x2.y; f.v[10] = (__bf16)x2.z; f.v[11] = (__bf16)x2.w;
  f.v[12] = (__bf16)x3.x; f.v[13] = (__bf16)x3.y; f.v[14] = (__bf16)x3.z; f.v[15] = (__bf16)x3.w;
  return f.v;
}

DEV v8f wmma_bf16(v16bf a, v16bf b, v8f c) {
  return __builtin_amdgcn_wmma_f32_16x16x32_bf16(false, a, false, b, (short)0, c, false, false);
}

// ---------------------------------------------------------------------------
// Small kernels: timestep embedding + label, conditioning GEMV, pos-embed
// ---------------------------------------------------------------------------

__global__ void embed_kernel(const float* __restrict__ t, const float* __restrict__ dt,
                             const int* __restrict__ y,
                             const float* __restrict__ te_w1, const float* __restrict__ te_b1,
                             const float* __restrict__ te_w2, const float* __restrict__ te_b2,
                             const float* __restrict__ dte_w1, const float* __restrict__ dte_b1,
                             const float* __restrict__ dte_w2, const float* __restrict__ dte_b2,
                             const float* __restrict__ label_emb,
                             float* __restrict__ c_out, float* __restrict__ sc_out) {
  int b = blockIdx.x;
  int tid = threadIdx.x; // 256 threads
  __shared__ float temb[256];
  __shared__ float h[HID];
  __shared__ float acc[HID];
  for (int o = tid; o < HID; o += 256) acc[o] = 0.f;
  for (int pass = 0; pass < 2; ++pass) {
    float tv = pass ? dt[b] : t[b];
    const float* w1 = pass ? dte_w1 : te_w1;
    const float* b1 = pass ? dte_b1 : te_b1;
    const float* w2 = pass ? dte_w2 : te_w2;
    const float* b2 = pass ? dte_b2 : te_b2;
    __syncthreads();
    {
      int i = tid & 127;
      float fr = __expf(-logf(10000.f) * (float)i / 128.f);
      float a = tv * fr;
      temb[tid] = (tid >= 128) ? sinf(a) : cosf(a); // cos first, then sin
    }
    __syncthreads();
    for (int o = tid; o < HID; o += 256) {
      float s = b1[o];
      const float* wr = w1 + (size_t)o * 256;
      for (int k = 0; k < 256; ++k) s += temb[k] * wr[k];
      h[o] = s / (1.f + __expf(-s)); // silu
    }
    __syncthreads();
    for (int o = tid; o < HID; o += 256) {
      float s = b2[o];
      const float* wr = w2 + (size_t)o * HID;
      for (int k = 0; k < HID; ++k) s += h[k] * wr[k];
      acc[o] += s;
    }
  }
  __syncthreads();
  int lab = y[b];
  for (int o = tid; o < HID; o += 256) {
    float cv = acc[o] + label_emb[(size_t)lab * HID + o];
    c_out[(size_t)b * HID + o]  = cv;
    sc_out[(size_t)b * HID + o] = cv / (1.f + __expf(-cv));
  }
}

// out[b][n] = sc[b] . W[n] + bias[n]  (W is [N x 768], sc is [16 x 768])
__global__ void cond_kernel(const float* __restrict__ sc, const float* __restrict__ W,
                            const float* __restrict__ bias, float* __restrict__ out, int N) {
  __shared__ float s[BATCH * HID];
  int tid = threadIdx.x;
  for (int i = tid; i < BATCH * HID; i += 256) s[i] = sc[i];
  __syncthreads();
  int n = blockIdx.x * 256 + tid;
  if (n >= N) return;
  float acc[BATCH];
  #pragma unroll
  for (int b = 0; b < BATCH; ++b) acc[b] = 0.f;
  const float* wr = W + (size_t)n * HID;
  for (int k = 0; k < HID; ++k) {
    float w = wr[k];
    #pragma unroll
    for (int b = 0; b < BATCH; ++b) acc[b] += s[b * HID + k] * w;
  }
  float bb = bias[n];
  #pragma unroll
  for (int b = 0; b < BATCH; ++b) out[(size_t)b * N + n] = acc[b] + bb;
}

// 2D sincos positional embedding, pos[1024][768]
__global__ void pos_embed_kernel(float* __restrict__ pos) {
  int i = blockIdx.x * 256 + threadIdx.x;
  if (i >= NTOK * HID) return;
  int t = i / HID;
  int k = i - t * HID;
  int half = k / 192;      // 0: sin(gw) 1: cos(gw) 2: sin(gh) 3: cos(gh)
  int kk = k - half * 192;
  float om = __powf(10000.f, -(float)kk / 192.f);
  float pv = (half < 2) ? (float)(t & 31) : (float)(t >> 5);
  float a = pv * om;
  pos[i] = (half & 1) ? cosf(a) : sinf(a);
}

// Gather strided conv patches into bf16 A matrix [16384 x 192], k=(c,p,q)
__global__ void patch_gather(const float* __restrict__ x, unsigned short* __restrict__ Ap) {
  int i = blockIdx.x * 256 + threadIdx.x;        // MTOT * 24
  if (i >= MTOT * 24) return;
  int m = i / 24;
  int cp = i - m * 24;
  int c = cp >> 3, pp = cp & 7;
  int b = m >> 10, tk = m & 1023;
  int gh = tk >> 5, gw = tk & 31;
  const float* src = x + ((((size_t)b * 3 + c) * 256 + gh * 8 + pp) * 256 + gw * 8);
  float4 v0 = *(const float4*)(src);
  float4 v1 = *(const float4*)(src + 4);
  unsigned short* dst = Ap + (size_t)m * PATCHK + c * 64 + pp * 8;
  dst[0] = f2bf(v0.x); dst[1] = f2bf(v0.y); dst[2] = f2bf(v0.z); dst[3] = f2bf(v0.w);
  dst[4] = f2bf(v1.x); dst[5] = f2bf(v1.y); dst[6] = f2bf(v1.z); dst[7] = f2bf(v1.w);
}

// ---------------------------------------------------------------------------
// LayerNorm + modulate -> bf16:  out = LN(x) * (1+scale[b]) + shift[b]
// ---------------------------------------------------------------------------
__global__ __launch_bounds__(256) void ln_mod(const float* __restrict__ x,
                                              const float* __restrict__ shift,
                                              const float* __restrict__ scale, int cstride,
                                              unsigned short* __restrict__ out) {
  int lane = threadIdx.x & 31;
  int wave = threadIdx.x >> 5;
  int row = blockIdx.x * 8 + wave;
  const float* xr = x + (size_t)row * HID;
  float vals[24];
  float s = 0.f, sq = 0.f;
  #pragma unroll
  for (int c = 0; c < 24; ++c) {
    float v = xr[c * 32 + lane];
    vals[c] = v; s += v; sq += v * v;
  }
  #pragma unroll
  for (int m = 1; m < 32; m <<= 1) { s += __shfl_xor(s, m, 32); sq += __shfl_xor(sq, m, 32); }
  float mean = s * (1.f / HID);
  float var = sq * (1.f / HID) - mean * mean;
  float inv = rsqrtf(var + 1e-5f);
  int b = row >> 10;
  const float* sh = shift + (size_t)b * cstride;
  const float* sc = scale + (size_t)b * cstride;
  unsigned short* orow = out + (size_t)row * HID;
  #pragma unroll
  for (int c = 0; c < 24; ++c) {
    int col = c * 32 + lane;
    float v = (vals[c] - mean) * inv;
    v = v * (1.f + sc[col]) + sh[col];
    orow[col] = f2bf(v);
  }
}

// ---------------------------------------------------------------------------
// bf16 WMMA GEMM: out[M,N] = epi( A[M,K](bf16) * W[N,K](f32)^T + bias )
// Wave tile 32x64 (2x4 WMMA tiles), 8 waves/WG -> WG tile 256x64, no LDS
// (weights/activations are L2-resident on the 192MB MALL).
// ---------------------------------------------------------------------------
enum { EPI_F32 = 0, EPI_POS = 1, EPI_BF16 = 2, EPI_BF16_T = 3, EPI_GELU = 4, EPI_RESGATE = 5 };

template <int EPI>
__global__ __launch_bounds__(256) void gemm_wmma(const unsigned short* __restrict__ A,
                                                 const float* __restrict__ W,
                                                 const float* __restrict__ bias, float alpha,
                                                 int M, int N, int K,
                                                 float* __restrict__ outF,
                                                 unsigned short* __restrict__ outB,
                                                 const float* __restrict__ extra, int estride) {
  int lane = threadIdx.x & 31;
  int wave = threadIdx.x >> 5;
  int m0 = blockIdx.x * 256 + wave * 32;
  int n0 = blockIdx.y * 64;

  v8f acc[2][4];
  #pragma unroll
  for (int i = 0; i < 2; ++i)
    #pragma unroll
    for (int j = 0; j < 4; ++j) acc[i][j] = v8f_zero();

  for (int kb = 0; kb < K; kb += 32) {
    // prefetch next k-step weight lines (global_prefetch_b8, no counter traffic)
    if (kb + 32 < K) {
      #pragma unroll
      for (int j = 0; j < 4; ++j)
        __builtin_prefetch(W + (size_t)(n0 + j * 16 + (lane & 15)) * K + kb + 32, 0, 0);
    }
    v16bf a0 = fragA_bf16(A, K, m0, kb, lane);
    v16bf a1 = fragA_bf16(A, K, m0 + 16, kb, lane);
    #pragma unroll
    for (int j = 0; j < 4; ++j) {
      v16bf bf = fragB_f32(W, K, n0 + j * 16, kb, lane);
      acc[0][j] = wmma_bf16(a0, bf, acc[0][j]);
      acc[1][j] = wmma_bf16(a1, bf, acc[1][j]);
    }
  }

  int hi = (lane >> 4) << 3;
  int cl = lane & 15;
  #pragma unroll
  for (int i = 0; i < 2; ++i) {
    #pragma unroll
    for (int j = 0; j < 4; ++j) {
      int col = n0 + j * 16 + cl;
      float bv = bias[col];
      #pragma unroll
      for (int r = 0; r < 8; ++r) {
        int row = m0 + i * 16 + hi + r;
        float v = alpha * (acc[i][j][r] + bv);
        if (EPI == EPI_F32) {
          outF[(size_t)row * N + col] = v;
        } else if (EPI == EPI_POS) {
          outF[(size_t)row * N + col] = v + extra[(size_t)(row & 1023) * N + col];
        } else if (EPI == EPI_BF16) {
          outB[(size_t)row * N + col] = f2bf(v);
        } else if (EPI == EPI_BF16_T) {
          int b = row >> 10, tk = row & 1023;   // -> V^T layout [b][n][tok]
          outB[(((size_t)(b * HID + col)) << 10) + tk] = f2bf(v);
        } else if (EPI == EPI_GELU) {
          float g = 0.5f * v * (1.f + erff(v * 0.70710678118f));
          outB[(size_t)row * N + col] = f2bf(g);
        } else { // EPI_RESGATE: x += gate[b,col] * v  (in place)
          int b = row >> 10;
          float gt = extra[(size_t)b * estride + col];
          float* o = outF + (size_t)row * N + col;
          *o = *o + gt * v;
        }
      }
    }
  }
}

// ---------------------------------------------------------------------------
// Flash attention: per-wave 16 query rows, online softmax over 32-key steps.
// Q,K bf16 [16384,768]; V^T bf16 [b][h*64+d][1024]; out bf16 [16384,768].
// Row max: DPP16 butterfly (VALU). Row sum: extra WMMA P @ ones (flash-v2
// style extra accumulator column) -> zero LDS reduction traffic.
// Only LDS use: per-wave P restage (C-layout -> A-layout).
// ---------------------------------------------------------------------------
__global__ __launch_bounds__(256) void attn_kernel(const unsigned short* __restrict__ Q,
                                                   const unsigned short* __restrict__ Kmat,
                                                   const unsigned short* __restrict__ Vt,
                                                   unsigned short* __restrict__ O) {
  __shared__ alignas(16) unsigned short Pst[8][16 * 40]; // 16 rows x 32 cols, stride 40
  int lane = threadIdx.x & 31;
  int wave = threadIdx.x >> 5;
  int bh = blockIdx.x;
  int b = bh / HEADS;
  int h = bh - b * HEADS;
  int q0 = blockIdx.y * 128 + wave * 16;
  int qrow = b * NTOK + q0;
  int hcol = h * HD;

  v16bf qf0 = fragA_bf16(Q, HID, qrow, hcol, lane);
  v16bf qf1 = fragA_bf16(Q, HID, qrow, hcol + 32, lane);

  v16bf ones;
  {
    FragU f;
    #pragma unroll
    for (int i = 0; i < 16; ++i) f.v[i] = (__bf16)1.0f;
    ones = f.v;
  }

  float mrow[8];
  #pragma unroll
  for (int r = 0; r < 8; ++r) mrow[r] = -1e30f;
  v8f lacc = v8f_zero();   // row-sum accumulator (every column = row sum)
  v8f oacc[4];
  #pragma unroll
  for (int j = 0; j < 4; ++j) oacc[j] = v8f_zero();

  const unsigned short* vbase = Vt + ((size_t)(b * HID + hcol)) * NTOK;
  unsigned short* ps = &Pst[wave][0];
  int hi = (lane >> 4) << 3;
  int cl = lane & 15;

  for (int kt = 0; kt < NTOK; kt += 32) {
    int krow = b * NTOK + kt;
    v8f s0 = v8f_zero(), s1 = v8f_zero();
    {
      v16bf k00 = fragB_bf16(Kmat, HID, krow, hcol, lane);
      v16bf k01 = fragB_bf16(Kmat, HID, krow, hcol + 32, lane);
      s0 = wmma_bf16(qf0, k00, s0);
      s0 = wmma_bf16(qf1, k01, s0);
      v16bf k10 = fragB_bf16(Kmat, HID, krow + 16, hcol, lane);
      v16bf k11 = fragB_bf16(Kmat, HID, krow + 16, hcol + 32, lane);
      s1 = wmma_bf16(qf0, k10, s1);
      s1 = wmma_bf16(qf1, k11, s1);
    }
    // online softmax update for the 8 rows this lane holds
    #pragma unroll
    for (int r = 0; r < 8; ++r) {
      float mx = wave16_max(fmaxf(s0[r], s1[r]));
      float nm = fmaxf(mrow[r], mx);
      float scl = __expf(mrow[r] - nm);
      mrow[r] = nm;
      float p0 = __expf(s0[r] - nm);
      float p1 = __expf(s1[r] - nm);
      lacc[r] *= scl;
      #pragma unroll
      for (int j = 0; j < 4; ++j) oacc[j][r] *= scl;
      int R = r + hi;
      ps[R * 40 + cl] = f2bf(p0);
      ps[R * 40 + 16 + cl] = f2bf(p1);
    }
    __syncthreads();
    v16bf pf = fragA_bf16(ps, 40, 0, 0, lane);
    lacc = wmma_bf16(pf, ones, lacc);  // row sums ride a WMMA, no shuffles
    #pragma unroll
    for (int j = 0; j < 4; ++j) {
      v16bf vf = fragB_bf16(vbase, NTOK, j * 16, kt, lane);
      oacc[j] = wmma_bf16(pf, vf, oacc[j]);
    }
    __syncthreads();
  }

  #pragma unroll
  for (int j = 0; j < 4; ++j) {
    #pragma unroll
    for (int r = 0; r < 8; ++r) {
      float val = oacc[j][r] / lacc[r];
      int row = qrow + hi + r;
      int col = hcol + j * 16 + cl;
      O[(size_t)row * HID + col] = f2bf(val);
    }
  }
}

// ---------------------------------------------------------------------------
// Unpatchify: fout[16384,192] -> out[16,3,256,256]
// ---------------------------------------------------------------------------
__global__ void unpatch(const float* __restrict__ f, float* __restrict__ out) {
  int i = blockIdx.x * 256 + threadIdx.x;
  if (i >= BATCH * 3 * 256 * 256) return;
  int ww = i & 255;
  int hh = (i >> 8) & 255;
  int bc = i >> 16;
  int b = bc / 3, c = bc - (bc / 3) * 3;
  int gh = hh >> 3, p = hh & 7, gw = ww >> 3, qq = ww & 7;
  int m = b * NTOK + gh * 32 + gw;
  int n = (p * 8 + qq) * 3 + c;
  out[i] = f[(size_t)m * PATCHK + n];
}

// ---------------------------------------------------------------------------
// Host driver
// ---------------------------------------------------------------------------
extern "C" void kernel_launch(void* const* d_in, const int* in_sizes, int n_in,
                              void* d_out, int out_size, void* d_ws, size_t ws_size,
                              hipStream_t stream) {
  (void)in_sizes; (void)n_in; (void)out_size; (void)ws_size;

  const float* x_in   = (const float*)d_in[0];
  const float* t_in   = (const float*)d_in[1];
  const float* dt_in  = (const float*)d_in[2];
  const int*   y_in   = (const int*)d_in[3];
  const float* patch_w = (const float*)d_in[4];
  const float* patch_b = (const float*)d_in[5];
  const float* te_w1 = (const float*)d_in[6];
  const float* te_b1 = (const float*)d_in[7];
  const float* te_w2 = (const float*)d_in[8];
  const float* te_b2 = (const float*)d_in[9];
  const float* dte_w1 = (const float*)d_in[10];
  const float* dte_b1 = (const float*)d_in[11];
  const float* dte_w2 = (const float*)d_in[12];
  const float* dte_b2 = (const float*)d_in[13];
  const float* label_emb = (const float*)d_in[14];
  const float* c_w  = (const float*)d_in[15];
  const float* c_b  = (const float*)d_in[16];
  const float* q_w  = (const float*)d_in[17];
  const float* q_b  = (const float*)d_in[18];
  const float* k_w  = (const float*)d_in[19];
  const float* k_b  = (const float*)d_in[20];
  const float* v_w  = (const float*)d_in[21];
  const float* v_b  = (const float*)d_in[22];
  const float* a_w  = (const float*)d_in[23];
  const float* a_b  = (const float*)d_in[24];
  const float* m_w1 = (const float*)d_in[25];
  const float* m_b1 = (const float*)d_in[26];
  const float* m_w2 = (const float*)d_in[27];
  const float* m_b2 = (const float*)d_in[28];
  const float* f_c_w = (const float*)d_in[29];
  const float* f_c_b = (const float*)d_in[30];
  const float* f_w   = (const float*)d_in[31];
  const float* f_b   = (const float*)d_in[32];

  // Workspace layout (bytes)
  const size_t SZ_X    = (size_t)MTOT * HID * 4;       // 50.3 MB f32 residual
  const size_t SZ_XMOD = (size_t)MTOT * HID * 2;       // 25.2 MB bf16
  const size_t SZ_BF   = (size_t)MTOT * HID * 2;       // one bf16 act tensor
  const size_t SZ_BIG  = 4 * SZ_BF;                    // == MTOT*MLPD*2
  char* W8 = (char*)d_ws;
  float*          xf    = (float*)(W8);
  unsigned short* xmod  = (unsigned short*)(W8 + SZ_X);
  char*           bigR  = W8 + SZ_X + SZ_XMOD;
  unsigned short* qb    = (unsigned short*)bigR;
  unsigned short* kbuf  = qb   + (size_t)MTOT * HID;
  unsigned short* vt    = kbuf + (size_t)MTOT * HID;
  unsigned short* attnb = vt   + (size_t)MTOT * HID;
  unsigned short* hb    = (unsigned short*)bigR;       // overlay (qkv dead)
  unsigned short* Ap    = (unsigned short*)bigR;       // overlay (start)
  float*          fout  = (float*)bigR;                // overlay (end)
  char*           smalls = bigR + SZ_BIG;
  float* pos   = (float*)(smalls);                     // 1024*768 f32
  float* cbuf  = (float*)(smalls + (size_t)NTOK * HID * 4);
  float* scbuf = cbuf  + (size_t)BATCH * HID;
  float* ccall = scbuf + (size_t)BATCH * HID;          // [12][16][4608]
  float* cc2   = ccall + (size_t)DEPTH * BATCH * CCN;  // [16][1536]

  // --- conditioning path ---
  embed_kernel<<<BATCH, 256, 0, stream>>>(t_in, dt_in, y_in, te_w1, te_b1, te_w2, te_b2,
                                          dte_w1, dte_b1, dte_w2, dte_b2, label_emb,
                                          cbuf, scbuf);
  for (int d = 0; d < DEPTH; ++d) {
    cond_kernel<<<CCN / 256, 256, 0, stream>>>(scbuf, c_w + (size_t)d * CCN * HID,
                                               c_b + (size_t)d * CCN,
                                               ccall + (size_t)d * BATCH * CCN, CCN);
  }
  cond_kernel<<<(2 * HID) / 256, 256, 0, stream>>>(scbuf, f_c_w, f_c_b, cc2, 2 * HID);

  // --- patch embed ---
  pos_embed_kernel<<<(NTOK * HID + 255) / 256, 256, 0, stream>>>(pos);
  patch_gather<<<(MTOT * 24 + 255) / 256, 256, 0, stream>>>(x_in, Ap);
  gemm_wmma<EPI_POS><<<dim3(MTOT / 256, HID / 64), 256, 0, stream>>>(
      Ap, patch_w, patch_b, 1.f, MTOT, HID, PATCHK, xf, nullptr, pos, 0);

  // --- transformer blocks ---
  for (int d = 0; d < DEPTH; ++d) {
    const float* ccd = ccall + (size_t)d * BATCH * CCN;
    const size_t wo = (size_t)d * HID * HID;
    const size_t mo = (size_t)d * MLPD * HID;

    ln_mod<<<MTOT / 8, 256, 0, stream>>>(xf, ccd + 0 * HID, ccd + 1 * HID, CCN, xmod);
    gemm_wmma<EPI_BF16><<<dim3(MTOT / 256, HID / 64), 256, 0, stream>>>(
        xmod, q_w + wo, q_b + (size_t)d * HID, 1.f / 64.f, MTOT, HID, HID, nullptr, qb, nullptr, 0);
    gemm_wmma<EPI_BF16><<<dim3(MTOT / 256, HID / 64), 256, 0, stream>>>(
        xmod, k_w + wo, k_b + (size_t)d * HID, 1.f, MTOT, HID, HID, nullptr, kbuf, nullptr, 0);
    gemm_wmma<EPI_BF16_T><<<dim3(MTOT / 256, HID / 64), 256, 0, stream>>>(
        xmod, v_w + wo, v_b + (size_t)d * HID, 1.f, MTOT, HID, HID, nullptr, vt, nullptr, 0);

    attn_kernel<<<dim3(BATCH * HEADS, NTOK / 128), 256, 0, stream>>>(qb, kbuf, vt, attnb);

    gemm_wmma<EPI_RESGATE><<<dim3(MTOT / 256, HID / 64), 256, 0, stream>>>(
        attnb, a_w + wo, a_b + (size_t)d * HID, 1.f, MTOT, HID, HID, xf, nullptr,
        ccd + 2 * HID, CCN);

    ln_mod<<<MTOT / 8, 256, 0, stream>>>(xf, ccd + 3 * HID, ccd + 4 * HID, CCN, xmod);
    gemm_wmma<EPI_GELU><<<dim3(MTOT / 256, MLPD / 64), 256, 0, stream>>>(
        xmod, m_w1 + mo, m_b1 + (size_t)d * MLPD, 1.f, MTOT, MLPD, HID, nullptr, hb, nullptr, 0);
    gemm_wmma<EPI_RESGATE><<<dim3(MTOT / 256, HID / 64), 256, 0, stream>>>(
        hb, m_w2 + mo, m_b2 + (size_t)d * HID, 1.f, MTOT, HID, MLPD, xf, nullptr,
        ccd + 5 * HID, CCN);
  }

  // --- final layer ---
  ln_mod<<<MTOT / 8, 256, 0, stream>>>(xf, cc2 + 0, cc2 + HID, 2 * HID, xmod);
  gemm_wmma<EPI_F32><<<dim3(MTOT / 256, PATCHK / 64), 256, 0, stream>>>(
      xmod, f_w, f_b, 1.f, MTOT, PATCHK, HID, fout, nullptr, nullptr, 0);
  unpatch<<<(BATCH * 3 * 256 * 256 + 255) / 256, 256, 0, stream>>>(fout, (float*)d_out);
}
